// NaiveAttention_47682726920223
// MI455X (gfx1250) — matrix operand order repaired and checked by reading, not compiled, not run on hardware
//
#include <hip/hip_runtime.h>
#include <hip/hip_bf16.h>

#define NUM_HEADS    32
#define NUM_KV_HEADS 8
#define HEAD_DIM     128
#define T_TOK        2048
#define S_TOK        2048

typedef __attribute__((ext_vector_type(16))) _Float16 v16h;
typedef __attribute__((ext_vector_type(8)))  _Float16 v8h;
typedef __attribute__((ext_vector_type(2)))  _Float16 h2;
typedef __attribute__((ext_vector_type(8)))  float    v8f;

#define KSTRIDE 136   // 128 + 8 f16 pad -> rotates LDS banks per row
#define VSTRIDE 40    // 32 + 8 f16 pad
#define PSTRIDE 40

union VFrag { v16h v; v8h h[2]; };

__launch_bounds__(128)
__global__ void fa_gqa_kernel(const float* __restrict__ q,
                              const float* __restrict__ k,
                              const float* __restrict__ v,
                              float* __restrict__ out)
{
    // double-buffered K/V chunks (software pipeline), per-wave P bounce
    __shared__ _Float16 ldsK[2][32 * KSTRIDE];         // K chunk, row-major f16
    __shared__ _Float16 ldsVt[2][HEAD_DIM * VSTRIDE];  // V chunk, transposed [d][s]
    __shared__ _Float16 ldsP[4][16 * PSTRIDE];

    const int tid  = threadIdx.x;
    const int wave = tid >> 5;
    const int lane = tid & 31;
    const int l    = lane & 15;   // column / M-row selector
    const int hh   = lane >> 4;   // half-wave

    const int qbase = blockIdx.x * 64;
    const int head  = blockIdx.y;
    const int kvh   = head >> 2;              // NUM_GROUPS == 4
    const int q0    = qbase + wave * 16;

    const size_t kvrow_stride = (size_t)NUM_KV_HEADS * HEAD_DIM;
    const float scaling = 0.08838834764831843f;  // HEAD_DIM^-0.5

    // ---- Q tile (16 x 128) -> 4 f16 A-fragments (16x32), scale folded in.
    // A layout: lane row M = l; element e -> K = hh*16 + e
    v16h aQ[4];
    {
        const float* qrow = q + (size_t)(q0 + l) * (NUM_HEADS * HEAD_DIM) + head * HEAD_DIM;
        #pragma unroll
        for (int c = 0; c < 4; ++c) {
            #pragma unroll
            for (int e = 0; e < 16; e += 2) {
                const int d = c * 32 + hh * 16 + e;
                float2 f = *(const float2*)(qrow + d);
                aQ[c][e]     = (_Float16)(f.x * scaling);
                aQ[c][e + 1] = (_Float16)(f.y * scaling);
            }
        }
    }

    v8f Oa[8] = {};                  // 16x128 fp32 output accumulators (8 C tiles)
    float m_s[8], l_s[8];            // online softmax state (rows r + hh*8)
    #pragma unroll
    for (int r = 0; r < 8; ++r) { m_s[r] = -1.0e30f; l_s[r] = 0.0f; }

    _Float16* ldsPw = ldsP[wave];

    const int tr = tid >> 2;          // staging: row within 32-row chunk
    const int tc = (tid & 3) * 32;    // staging: 32-col segment base

    const int nchunks = (qbase + 64) / 32;   // causal bound for this 64-row block

    // ---- prologue: stage chunk 0 into buffer 0
    {
        const float* krow = k + (size_t)tr * kvrow_stride + kvh * HEAD_DIM + tc;
        const float* vrow = v + (size_t)tr * kvrow_stride + kvh * HEAD_DIM + tc;
        _Float16* dst = &ldsK[0][tr * KSTRIDE + tc];
        #pragma unroll
        for (int i = 0; i < 16; ++i) {
            float2 f = *(const float2*)(krow + 2 * i);
            h2 hv; hv.x = (_Float16)f.x; hv.y = (_Float16)f.y;
            *(h2*)(dst + 2 * i) = hv;
            float2 g = *(const float2*)(vrow + 2 * i);
            ldsVt[0][(tc + 2 * i)     * VSTRIDE + tr] = (_Float16)g.x;
            ldsVt[0][(tc + 2 * i + 1) * VSTRIDE + tr] = (_Float16)g.y;
        }
    }
    __syncthreads();

    for (int jc = 0; jc < nchunks; ++jc) {
        const int j   = jc * 32;
        const int buf = jc & 1;
        const bool havenext = (jc + 1 < nchunks);

        // ---- issue global loads + converts for chunk jc+1 (latency hides under compute)
        h2 kreg[16], vreg[16];
        if (havenext) {
            const float* krow = k + (size_t)(j + 32 + tr) * kvrow_stride + kvh * HEAD_DIM + tc;
            const float* vrow = v + (size_t)(j + 32 + tr) * kvrow_stride + kvh * HEAD_DIM + tc;
            #pragma unroll
            for (int i = 0; i < 16; ++i) {
                float2 f = *(const float2*)(krow + 2 * i);
                kreg[i].x = (_Float16)f.x; kreg[i].y = (_Float16)f.y;
                float2 g = *(const float2*)(vrow + 2 * i);
                vreg[i].x = (_Float16)g.x; vreg[i].y = (_Float16)g.y;
            }
        }
        // speculative prefetch of the chunk after next toward L2/WGP$
        if (jc + 2 < nchunks) {
            __builtin_prefetch(k + (size_t)(j + 64 + tr) * kvrow_stride + kvh * HEAD_DIM + tc, 0, 1);
            __builtin_prefetch(v + (size_t)(j + 64 + tr) * kvrow_stride + kvh * HEAD_DIM + tc, 0, 1);
        }

        const _Float16* Kb  = ldsK[buf];
        const _Float16* Vtb = ldsVt[buf];

        // ---- scores: S0 = Q x K^T cols [j, j+16), S1 = cols [j+16, j+32)
        // B layout (32x16): lane col N = l, element e -> K = hh*16 + e
        v8f S0 = {}, S1 = {};
        #pragma unroll
        for (int c = 0; c < 4; ++c) {
            VFrag b0, b1;
            const _Float16* base0 = &Kb[l * KSTRIDE + c * 32 + hh * 16];
            const _Float16* base1 = &Kb[(16 + l) * KSTRIDE + c * 32 + hh * 16];
            b0.h[0] = *(const v8h*)(base0);
            b0.h[1] = *(const v8h*)(base0 + 8);
            b1.h[0] = *(const v8h*)(base1);
            b1.h[1] = *(const v8h*)(base1 + 8);
            S0 = __builtin_amdgcn_wmma_f32_16x16x32_f16(false, aQ[c], false, b0.v,
                                                        (short)0, S0, false, false);
            S1 = __builtin_amdgcn_wmma_f32_16x16x32_f16(false, aQ[c], false, b1.v,
                                                        (short)0, S1, false, false);
        }

        // ---- online softmax update (C layout: element r -> row q0 + r + hh*8, col l)
        float corr[8];
        #pragma unroll
        for (int r = 0; r < 8; ++r) {
            const int trow = q0 + r + hh * 8;
            float s0 = S0[r], s1 = S1[r];
            s0 = ((j + l)      > trow) ? -1.0e30f : s0;   // causal mask (select, no branch)
            s1 = ((j + 16 + l) > trow) ? -1.0e30f : s1;
            float rm = fmaxf(s0, s1);
            #pragma unroll
            for (int off = 1; off < 16; off <<= 1)
                rm = fmaxf(rm, __shfl_xor(rm, off, 32));
            const float mnew = fmaxf(m_s[r], rm);
            corr[r] = __expf(m_s[r] - mnew);
            const float p0 = __expf(s0 - mnew);
            const float p1 = __expf(s1 - mnew);
            float rs = p0 + p1;
            #pragma unroll
            for (int off = 1; off < 16; off <<= 1)
                rs += __shfl_xor(rs, off, 32);
            l_s[r] = l_s[r] * corr[r] + rs;
            m_s[r] = mnew;
            // bounce P through per-wave LDS to convert C layout -> A layout
            ldsPw[(r + hh * 8) * PSTRIDE + l]      = (_Float16)p0;
            ldsPw[(r + hh * 8) * PSTRIDE + 16 + l] = (_Float16)p1;
        }

        // rescale existing output accumulators
        #pragma unroll
        for (int t = 0; t < 8; ++t)
            #pragma unroll
            for (int r = 0; r < 8; ++r)
                Oa[t][r] *= corr[r];

        // ---- gather P as 16x32 f16 A-fragment
        v16h aP;
        #pragma unroll
        for (int e = 0; e < 16; e += 2) {
            const int kk = hh * 16 + e;
            h2 pv = *(const h2*)&ldsPw[l * PSTRIDE + kk];
            aP[e] = pv.x; aP[e + 1] = pv.y;
        }

        // ---- O += P x V  (V B-fragment from transposed LDS: contiguous reads)
        #pragma unroll
        for (int t = 0; t < 8; ++t) {
            VFrag bv;
            const _Float16* base = &Vtb[(t * 16 + l) * VSTRIDE + hh * 16];
            bv.h[0] = *(const v8h*)(base);
            bv.h[1] = *(const v8h*)(base + 8);
            Oa[t] = __builtin_amdgcn_wmma_f32_16x16x32_f16(false, aP, false, bv.v,
                                                           (short)0, Oa[t], false, false);
        }

        // ---- store staged chunk jc+1 into the other buffer (overlapped with compute above)
        if (havenext) {
            _Float16* dst = &ldsK[buf ^ 1][tr * KSTRIDE + tc];
            #pragma unroll
            for (int i = 0; i < 16; ++i)
                *(h2*)(dst + 2 * i) = kreg[i];
            _Float16* vb = ldsVt[buf ^ 1];
            #pragma unroll
            for (int i = 0; i < 16; ++i) {
                vb[(tc + 2 * i)     * VSTRIDE + tr] = vreg[i].x;
                vb[(tc + 2 * i + 1) * VSTRIDE + tr] = vreg[i].y;
            }
        }
        __syncthreads();   // chunk jc+1 visible; buffer (jc&1) free for chunk jc+2
    }

    // ---- normalize and store: out[t, head, d] fp32
    float inv[8];
    #pragma unroll
    for (int r = 0; r < 8; ++r) inv[r] = 1.0f / l_s[r];
    #pragma unroll
    for (int t = 0; t < 8; ++t) {
        #pragma unroll
        for (int r = 0; r < 8; ++r) {
            const int trow = q0 + r + hh * 8;
            out[(size_t)trow * (NUM_HEADS * HEAD_DIM) + head * HEAD_DIM + t * 16 + l]
                = Oa[t][r] * inv[r];
        }
    }
}

extern "C" void kernel_launch(void* const* d_in, const int* in_sizes, int n_in,
                              void* d_out, int out_size, void* d_ws, size_t ws_size,
                              hipStream_t stream) {
    const float* q = (const float*)d_in[0];
    const float* k = (const float*)d_in[1];
    const float* v = (const float*)d_in[2];
    float* out = (float*)d_out;
    dim3 grid(T_TOK / 64, NUM_HEADS);
    fa_gqa_kernel<<<grid, dim3(128), 0, stream>>>(q, k, v, out);
}
